// MAB_46823733461524
// MI455X (gfx1250) — compile-verified, matrix operand order checked by reading, not run on hardware
//
#include <hip/hip_runtime.h>
#include <hip/hip_bf16.h>

// ---------------------------------------------------------------------------
// MAB (multihead attention block) for MI455X / gfx1250, wave32 + WMMA f16->f32
// Register-blocked GEMMs: one wave computes a 32x64 tile (2x4 WMMA tiles).
// ---------------------------------------------------------------------------

typedef __attribute__((ext_vector_type(16))) _Float16 v16h;
typedef __attribute__((ext_vector_type(8)))  _Float16 v8h;
typedef __attribute__((ext_vector_type(8)))  float    v8f;

#define DIMV   512
#define ROWS   8192          // B*N = 8*1024
#define SEQ    1024
#define NHEAD  8
#define HDIM   64
#define PROB_LD 1032         // f16 LDS row stride (pad: 2064B = 516 dwords)

__device__ __forceinline__ v8f wmma_f16(v16h a, v16h b, v8f c) {
  return __builtin_amdgcn_wmma_f32_16x16x32_f16(false, a, false, b,
                                                (short)0, c, false, false);
}

// Load a 16x32 fragment. `base` points at (row0, k0) of a row-major [rows x ld]
// f16 matrix. Lane r = lane&15 supplies row r; g = lane>>4 selects K-half.
// Elements i=0..7 -> K = g*8+i ; i=8..15 -> K = 16+g*8+(i-8)  (ISA 7.12.2).
// Works for A fragments (rows = M) and B fragments when B is stored
// column-major (rows = N, i.e. pass B^T row-major).
__device__ __forceinline__ v16h load_frag(const _Float16* __restrict__ base, int ld) {
  int lane = threadIdx.x & 31;
  int r = lane & 15;
  int g = lane >> 4;
  const _Float16* p = base + r * ld + g * 8;
  union { v16h v; v8h h[2]; } u;
  u.h[0] = *(const v8h*)(p);
  u.h[1] = *(const v8h*)(p + 16);
  return u.v;
}

// ---------------------------------------------------------------------------
// 1) f32 -> f16 copy
__global__ void cvt_f16_kernel(const float* __restrict__ src,
                               _Float16* __restrict__ dst, int n) {
  int i = blockIdx.x * blockDim.x + threadIdx.x;
  if (i < n) dst[i] = (_Float16)src[i];
}

// 2) transpose 512x512 f32 -> f16  (dst[n*512+k] = src[k*512+n])
__global__ void cvt_wT_kernel(const float* __restrict__ src,
                              _Float16* __restrict__ dst) {
  int o = blockIdx.x * blockDim.x + threadIdx.x;
  if (o >= DIMV * DIMV) return;
  int n = o / DIMV, k = o % DIMV;
  dst[o] = (_Float16)src[k * DIMV + n];
}

// ---------------------------------------------------------------------------
// 3) Fused projections: Qp = Q@Wq+bq (f32+f16), Kp = K@Wk+bk (f16),
//    Vp = K@Wv+bv stored transposed [DIMV][ROWS] (f16).
//    One wave per 32x64 output tile: waves = 3 * (8192/32) * (512/64) = 6144.
__global__ __launch_bounds__(256) void proj_kernel(
    const _Float16* __restrict__ q16, const _Float16* __restrict__ k16,
    const _Float16* __restrict__ w16T,
    const float* __restrict__ bq, const float* __restrict__ bk,
    const float* __restrict__ bvv,
    float* __restrict__ qp32, _Float16* __restrict__ qp16,
    _Float16* __restrict__ kp16, _Float16* __restrict__ vpT) {
  int wave = (blockIdx.x * blockDim.x + threadIdx.x) >> 5;
  int p = wave / 2048;             // 0:Q 1:K 2:V  (2048 waves each)
  int t = wave % 2048;             // 256 row-groups x 8 col-groups
  int row0 = (t >> 3) * 32;
  int col0 = (t & 7) * 64;

  const _Float16* X = (p == 0) ? q16 : k16;
  const _Float16* abase = X + (size_t)row0 * DIMV;
  const _Float16* bbase = w16T + (size_t)p * (DIMV * DIMV) + (size_t)col0 * DIMV;

  v8f acc[2][4] = {};
  for (int k0 = 0; k0 < DIMV; k0 += 32) {
    v16h a0 = load_frag(abase + k0, DIMV);
    v16h a1 = load_frag(abase + 16 * DIMV + k0, DIMV);
#pragma unroll
    for (int j = 0; j < 4; ++j) {
      v16h b = load_frag(bbase + j * 16 * DIMV + k0, DIMV);
      acc[0][j] = wmma_f16(a0, b, acc[0][j]);
      acc[1][j] = wmma_f16(a1, b, acc[1][j]);
    }
  }

  int lane = threadIdx.x & 31;
  int n = lane & 15, mb = (lane >> 4) * 8;
  const float* bias = (p == 0) ? bq : (p == 1) ? bk : bvv;
#pragma unroll
  for (int j = 0; j < 4; ++j) {
    float bval = bias[col0 + j * 16 + n];
#pragma unroll
    for (int i = 0; i < 2; ++i) {
      for (int v = 0; v < 8; ++v) {
        int r = row0 + i * 16 + mb + v;
        int c = col0 + j * 16 + n;
        float val = acc[i][j][v] + bval;
        if (p == 0) {
          qp32[(size_t)r * DIMV + c] = val;
          qp16[(size_t)r * DIMV + c] = (_Float16)val;
        } else if (p == 1) {
          kp16[(size_t)r * DIMV + c] = (_Float16)val;
        } else {
          vpT[(size_t)c * ROWS + r] = (_Float16)val;
        }
      }
    }
  }
}

// ---------------------------------------------------------------------------
// 4) Attention per (b, h, 16-query block): scores -> softmax -> P@V + residual.
//    4 waves / block; wave w owns keys [w*256, w*256+256) for scores and
//    head-dim cols [w*16, w*16+16) for the output GEMM.
__global__ __launch_bounds__(128) void attn_kernel(
    const _Float16* __restrict__ qp16, const _Float16* __restrict__ kp16,
    const _Float16* __restrict__ vpT, const float* __restrict__ qp32,
    float* __restrict__ ao32) {
  __shared__ _Float16 pr[16 * PROB_LD];   // scores then exp(scores), f16
  __shared__ float red[16 * 8];
  __shared__ float rowmax[16];
  __shared__ float rowinv[16];

  int bid = blockIdx.x;
  int qBlk = bid & 63;
  int h = (bid >> 6) & 7;
  int b = bid >> 9;
  int grow = b * SEQ + (qBlk << 4);

  int tid = threadIdx.x;
  int w = tid >> 5, lane = tid & 31;
  int n = lane & 15, mb = (lane >> 4) * 8;

  // ---- scores: S = (Qh @ Kh^T) / 8, stored f16 in LDS
  const _Float16* qbase = qp16 + (size_t)grow * DIMV + h * HDIM;
  v16h qa0 = load_frag(qbase, DIMV);       // loop-invariant Q fragments
  v16h qa1 = load_frag(qbase + 32, DIMV);
  for (int j = 0; j < 16; ++j) {
    int n0 = w * 256 + j * 16;
    const _Float16* kbase = kp16 + (size_t)(b * SEQ + n0) * DIMV + h * HDIM;
    v8f acc = {};
    acc = wmma_f16(qa0, load_frag(kbase, DIMV), acc);
    acc = wmma_f16(qa1, load_frag(kbase + 32, DIMV), acc);
    for (int v = 0; v < 8; ++v)
      pr[(mb + v) * PROB_LD + n0 + n] = (_Float16)(acc[v] * 0.125f);
  }
  __syncthreads();

  // ---- softmax (two pass; defer the 1/sum to the output scale)
  int r = tid & 15, seg = tid >> 4;            // 8 segments x 128 keys
  {
    float mx = -3.0e38f;
    const _Float16* srow = pr + r * PROB_LD + seg * 128;
    for (int c = 0; c < 128; ++c) mx = fmaxf(mx, (float)srow[c]);
    red[r * 8 + seg] = mx;
  }
  __syncthreads();
  if (tid < 16) {
    float mx = red[tid * 8];
    for (int s = 1; s < 8; ++s) mx = fmaxf(mx, red[tid * 8 + s]);
    rowmax[tid] = mx;
  }
  __syncthreads();
  {
    float mx = rowmax[r];
    float sum = 0.f;
    _Float16* prow = pr + r * PROB_LD + seg * 128;
    for (int c = 0; c < 128; ++c) {
      float e = __expf((float)prow[c] - mx);
      sum += e;
      prow[c] = (_Float16)e;
    }
    red[r * 8 + seg] = sum;
  }
  __syncthreads();
  if (tid < 16) {
    float s = 0.f;
    for (int i = 0; i < 8; ++i) s += red[tid * 8 + i];
    rowinv[tid] = 1.0f / s;
  }
  __syncthreads();

  // ---- O = (P @ V) * rowinv + Qp   (wave w -> head-dim cols [w*16, w*16+16))
  int d0 = w * 16;
  const _Float16* vb = vpT + (size_t)(h * HDIM + d0) * ROWS + b * SEQ;
  v8f acc = {};
  for (int kk = 0; kk < SEQ; kk += 32) {
    v16h a = load_frag(pr + kk, PROB_LD);
    v16h bf = load_frag(vb + kk, ROWS);
    acc = wmma_f16(a, bf, acc);
  }
  for (int v = 0; v < 8; ++v) {
    int m = mb + v;
    size_t idx = (size_t)(grow + m) * DIMV + h * HDIM + d0 + n;
    ao32[idx] = acc[v] * rowinv[m] + qp32[idx];
  }
}

// ---------------------------------------------------------------------------
// 5) LayerNorm per row (wave per row). Optionally emits an f16 copy.
__global__ __launch_bounds__(256) void ln_kernel(
    const float* __restrict__ in, float* __restrict__ out32,
    _Float16* __restrict__ out16, const float* __restrict__ gamma,
    const float* __restrict__ beta) {
  int w = threadIdx.x >> 5, lane = threadIdx.x & 31;
  int row = blockIdx.x * 8 + w;
  const float* base = in + (size_t)row * DIMV;

  float x[16], sum = 0.f, sq = 0.f;
  for (int i = 0; i < 16; ++i) {
    x[i] = base[lane + i * 32];
    sum += x[i];
    sq += x[i] * x[i];
  }
  for (int off = 16; off > 0; off >>= 1) {
    sum += __shfl_xor(sum, off, 32);
    sq  += __shfl_xor(sq,  off, 32);
  }
  float mean = sum * (1.0f / DIMV);
  float var = sq * (1.0f / DIMV) - mean * mean;
  float rstd = rsqrtf(var + 1e-5f);
  for (int i = 0; i < 16; ++i) {
    int c = lane + i * 32;
    float y = (x[i] - mean) * rstd * gamma[c] + beta[c];
    out32[(size_t)row * DIMV + c] = y;
    if (out16) out16[(size_t)row * DIMV + c] = (_Float16)y;
  }
}

// ---------------------------------------------------------------------------
// 6) Out = X1 + relu(X1 @ Wo + bo). One wave per 32x64 tile, waves = 2048.
__global__ __launch_bounds__(256) void gemm2_kernel(
    const _Float16* __restrict__ x16, const _Float16* __restrict__ woT,
    const float* __restrict__ bo, const float* __restrict__ x32,
    float* __restrict__ out) {
  int wave = (blockIdx.x * blockDim.x + threadIdx.x) >> 5;
  int row0 = (wave >> 3) * 32;     // 256 row-groups
  int col0 = (wave & 7) * 64;      // 8 col-groups
  const _Float16* abase = x16 + (size_t)row0 * DIMV;
  const _Float16* bbase = woT + (size_t)col0 * DIMV;

  v8f acc[2][4] = {};
  for (int k0 = 0; k0 < DIMV; k0 += 32) {
    v16h a0 = load_frag(abase + k0, DIMV);
    v16h a1 = load_frag(abase + 16 * DIMV + k0, DIMV);
#pragma unroll
    for (int j = 0; j < 4; ++j) {
      v16h b = load_frag(bbase + j * 16 * DIMV + k0, DIMV);
      acc[0][j] = wmma_f16(a0, b, acc[0][j]);
      acc[1][j] = wmma_f16(a1, b, acc[1][j]);
    }
  }

  int lane = threadIdx.x & 31;
  int n = lane & 15, mb = (lane >> 4) * 8;
#pragma unroll
  for (int j = 0; j < 4; ++j) {
    float bval = bo[col0 + j * 16 + n];
#pragma unroll
    for (int i = 0; i < 2; ++i) {
      for (int v = 0; v < 8; ++v) {
        size_t idx = (size_t)(row0 + i * 16 + mb + v) * DIMV + col0 + j * 16 + n;
        out[idx] = x32[idx] + fmaxf(acc[i][j][v] + bval, 0.0f);
      }
    }
  }
}

// ---------------------------------------------------------------------------
extern "C" void kernel_launch(void* const* d_in, const int* in_sizes, int n_in,
                              void* d_out, int out_size, void* d_ws, size_t ws_size,
                              hipStream_t stream) {
  const float* Q  = (const float*)d_in[0];
  const float* K  = (const float*)d_in[1];
  const float* Wq = (const float*)d_in[2];
  const float* bq = (const float*)d_in[3];
  const float* Wk = (const float*)d_in[4];
  const float* bk = (const float*)d_in[5];
  const float* Wv = (const float*)d_in[6];
  const float* bv = (const float*)d_in[7];
  const float* Wo = (const float*)d_in[8];
  const float* bo = (const float*)d_in[9];
  const float* g0 = (const float*)d_in[10];
  const float* b0 = (const float*)d_in[11];
  const float* g1 = (const float*)d_in[12];
  const float* b1 = (const float*)d_in[13];
  float* out = (float*)d_out;

  char* ws = (char*)d_ws;
  const size_t NE = (size_t)ROWS * DIMV;      // 4,194,304 elements
  _Float16* q16  = (_Float16*)(ws);                         // 8 MB
  _Float16* k16  = (_Float16*)(ws + 8388608);               // 8 MB
  _Float16* w16T = (_Float16*)(ws + 16777216);              // 2 MB (4 mats)
  float*    qp32 = (float*)   (ws + 18874368);              // 16 MB
  _Float16* qp16 = (_Float16*)(ws + 35651584);              // 8 MB
  _Float16* kp16 = (_Float16*)(ws + 44040192);              // 8 MB
  _Float16* vpT  = (_Float16*)(ws + 52428800);              // 8 MB
  float*    ao32 = (float*)   (ws + 60817408);              // 16 MB
  float*    x132 = (float*)   (ws + 77594624);              // 16 MB
  _Float16* x116 = (_Float16*)(ws + 94371840);              // 8 MB
  // total workspace: ~98 MB

  // 1) convert activations
  cvt_f16_kernel<<<(int)(NE / 256), 256, 0, stream>>>(Q, q16, (int)NE);
  cvt_f16_kernel<<<(int)(NE / 256), 256, 0, stream>>>(K, k16, (int)NE);

  // 2) transposed f16 weights
  const int WN = DIMV * DIMV;
  cvt_wT_kernel<<<WN / 256, 256, 0, stream>>>(Wq, w16T + 0 * WN);
  cvt_wT_kernel<<<WN / 256, 256, 0, stream>>>(Wk, w16T + 1 * WN);
  cvt_wT_kernel<<<WN / 256, 256, 0, stream>>>(Wv, w16T + 2 * WN);
  cvt_wT_kernel<<<WN / 256, 256, 0, stream>>>(Wo, w16T + 3 * WN);

  // 3) projections: 6144 tile-waves (32x64 tiles), 8 waves/block
  proj_kernel<<<768, 256, 0, stream>>>(q16, k16, w16T, bq, bk, bv,
                                       qp32, qp16, kp16, vpT);

  // 4) attention: B*H*(SEQ/16) = 4096 blocks of 128 threads
  attn_kernel<<<4096, 128, 0, stream>>>(qp16, kp16, vpT, qp32, ao32);

  // 5) LN0 -> x1 (f32 + f16)
  ln_kernel<<<ROWS / 8, 256, 0, stream>>>(ao32, x132, x116, g0, b0);

  // 6) x1 + relu(x1 @ Wo + bo) -> d_out ; 2048 tile-waves
  gemm2_kernel<<<256, 256, 0, stream>>>(x116, w16T + 3 * WN, bo, x132, out);

  // 7) LN1 in place on d_out
  ln_kernel<<<ROWS / 8, 256, 0, stream>>>(out, out, (_Float16*)nullptr, g1, b1);
}
